// KVMemoryInjector_20615843021105
// MI455X (gfx1250) — matrix operand order repaired and checked by reading, not compiled, not run on hardware
//
#include <hip/hip_runtime.h>
#include <hip/hip_bf16.h>

// ---------------- problem constants (from reference) ----------------
constexpr int Tq   = 2048;          // tokens
constexpr int Dm   = 4096;          // model dim
constexpr int NH   = 32;            // query heads
constexpr int NKV  = 8;             // kv heads
constexpr int HDs  = 128;          // head dim
constexpr int RV   = 128;          // virtual kv tokens
constexpr int STOT = RV + Tq;      // 2176 total kv length
constexpr int NREP = NH / NKV;     // 4

typedef __bf16 bf_t;
typedef __attribute__((ext_vector_type(16))) __bf16 v16bf;
typedef __attribute__((ext_vector_type(8)))  __bf16 v8bf;
typedef __attribute__((ext_vector_type(8)))  float  v8f;
typedef __attribute__((ext_vector_type(4)))  unsigned uint4v;
typedef __attribute__((ext_vector_type(8)))  int      int8v;
typedef __attribute__((ext_vector_type(4)))  int      int4v;

// CDNA5 data-mover availability
#if defined(__has_builtin)
#if __has_builtin(__builtin_amdgcn_tensor_load_to_lds)
#define HAS_TDM_BUILTIN 1
#endif
#endif
#ifndef HAS_TDM_BUILTIN
#define HAS_TDM_BUILTIN 0
#endif
#define USE_ASYNC 1

__device__ inline v8f zero8() {
  v8f z;
  #pragma unroll
  for (int i = 0; i < 8; i++) z[i] = 0.0f;
  return z;
}

__device__ inline v16bf load2(const bf_t* p0, const bf_t* p1) {
  union { v16bf v; v8bf h[2]; } u;
  u.h[0] = *(const v8bf*)p0;
  u.h[1] = *(const v8bf*)p1;
  return u.v;
}

__device__ inline v8f wmma_bf16(v16bf a, v16bf b, v8f c) {
  // D = A(16x32 bf16) * B(32x16 bf16) + C(16x16 f32)
  return __builtin_amdgcn_wmma_f32_16x16x32_bf16(
      /*neg_a=*/false, a, /*neg_b=*/false, b,
      /*c_mod=*/(short)0, c, /*reuse_a=*/false, /*reuse_b=*/false);
}

// async copy: 16B per lane, global -> LDS, ASYNCcnt-tracked (no VGPR data path)
__device__ inline void async_b128(unsigned lds_dst, const void* gsrc) {
  unsigned long long ga = (unsigned long long)(size_t)gsrc;
  asm volatile("global_load_async_to_lds_b128 %0, %1, off"
               :: "v"(lds_dst), "v"(ga) : "memory");
}
__device__ inline void async_b128_off16(unsigned lds_dst, const void* gsrc) {
  unsigned long long ga = (unsigned long long)(size_t)gsrc;
  asm volatile("global_load_async_to_lds_b128 %0, %1, off offset:16"
               :: "v"(lds_dst), "v"(ga) : "memory");
}
__device__ inline void wait_async0() {
  asm volatile("s_wait_asynccnt 0x0" ::: "memory");
}

// TDM: issue a 2D tensor tile DMA (D# groups 0 and 1; VADDR2/3 disabled)
__device__ inline void tdm_load(uint4v g0, int8v g1) {
#if HAS_TDM_BUILTIN
  int4v gz;
  gz[0] = gz[1] = gz[2] = gz[3] = 0;
#if __clang_major__ >= 23
  int8v gz8;
  #pragma unroll
  for (int e = 0; e < 8; e++) gz8[e] = 0;
  __builtin_amdgcn_tensor_load_to_lds(g0, g1, gz, gz, gz8, 0);
#else
  __builtin_amdgcn_tensor_load_to_lds(g0, g1, gz, gz, 0);
#endif
#else
  // portable path: bypass the clang builtin (arity differs across toolchains)
  asm volatile("tensor_load_to_lds %0, %1" :: "s"(g0), "s"(g1) : "memory");
#endif
}
__device__ inline void wait_tensor0() {
#if defined(__has_builtin)
#if __has_builtin(__builtin_amdgcn_s_wait_tensorcnt)
  __builtin_amdgcn_s_wait_tensorcnt(0);
#else
  asm volatile("s_wait_tensorcnt 0x0" ::: "memory");
#endif
#else
  asm volatile("s_wait_tensorcnt 0x0" ::: "memory");
#endif
}

// ---------------- f32 -> bf16 cast ----------------
__global__ void cast_f32_bf16(const float* __restrict__ in, bf_t* __restrict__ out, long n) {
  long i = (long)blockIdx.x * blockDim.x + threadIdx.x;
  long stride = (long)gridDim.x * blockDim.x;
  for (; i < n; i += stride) out[i] = (bf_t)in[i];
}

// ---------------- bf16 WMMA GEMM: C[f32, MxN] = A[bf16, MxK] * B[bf16, KxN] ----------------
// Block tile 128x128x32, 256 threads = 8 waves in a 2x4 wave grid (64x32 per wave).
constexpr int LDA_S = 40;   // padded LDS strides (bf16 elems), rows stay 16B aligned
constexpr int LDB_S = 40;

__global__ __launch_bounds__(256) void gemm_bf16(
    const bf_t* __restrict__ A, const bf_t* __restrict__ B, float* __restrict__ C,
    int M, int N, int K) {
  __shared__ alignas(16) bf_t As[128 * LDA_S];       // [row][k]
  __shared__ alignas(16) bf_t Bs[128 * LDB_S];       // transposed: [col][k]
  const int tid   = threadIdx.x;
  const int lane  = tid & 31;
  const int wave  = tid >> 5;
  const int waveM = wave >> 2;                       // 0..1
  const int waveN = wave & 3;                        // 0..3
  const int bRow  = blockIdx.y * 128;
  const int bCol  = blockIdx.x * 128;
  const int lc    = lane & 15;
  const int half  = lane >> 4;

  v8f acc[4][2];
  #pragma unroll
  for (int i = 0; i < 4; i++)
    #pragma unroll
    for (int j = 0; j < 2; j++) acc[i][j] = zero8();

  const int arow = tid >> 1, ahalf = tid & 1;        // A loader: 128 rows x 2 half-rows
  const int brow = tid >> 3, bseg  = tid & 7;        // B loader: 32 rows x 8 col-segments
  const unsigned As_dst = (unsigned)(size_t)(void*)(As + arow * LDA_S + ahalf * 16);

  for (int k0 = 0; k0 < K; k0 += 32) {
    { // stage A tile 128x32 (no transpose -> async DMA to LDS)
      const bf_t* src = A + (size_t)(bRow + arow) * K + k0 + ahalf * 16;
#if USE_ASYNC
      async_b128(As_dst, src);           // bytes [0,16): same +off on LDS & global
      async_b128_off16(As_dst, src);     // bytes [16,32)
#else
      bf_t* dst = As + arow * LDA_S + ahalf * 16;
      *(v8bf*)dst       = *(const v8bf*)src;
      *(v8bf*)(dst + 8) = *(const v8bf*)(src + 8);
#endif
      if (k0 + 32 < K) __builtin_prefetch(src + 32, 0, 1);   // global_prefetch_b8
    }
    { // stage B tile 32x128, transposed into [col][k]
      const bf_t* src = B + (size_t)(k0 + brow) * N + bCol + bseg * 16;
      v8bf b0 = *(const v8bf*)src;
      v8bf b1 = *(const v8bf*)(src + 8);
      #pragma unroll
      for (int i = 0; i < 8; i++) Bs[(bseg * 16 + i) * LDB_S + brow] = b0[i];
      #pragma unroll
      for (int i = 0; i < 8; i++) Bs[(bseg * 16 + 8 + i) * LDB_S + brow] = b1[i];
    }
#if USE_ASYNC
    wait_async0();
#endif
    __syncthreads();

    v16bf afr[4], bfr[2];
    #pragma unroll
    for (int mt = 0; mt < 4; mt++) {
      const int m = waveM * 64 + mt * 16 + lc;       // A lane row = lane%16
      // A fragment: K runs [8*half, +8) and [16+8*half, +8)
      afr[mt] = load2(&As[m * LDA_S + 8 * half], &As[m * LDA_S + 16 + 8 * half]);
    }
    #pragma unroll
    for (int nt = 0; nt < 2; nt++) {
      const int n = waveN * 32 + nt * 16 + lc;       // B lane col = lane%16
      // B fragment: K runs [8*half, +8) and [16+8*half, +8)
      bfr[nt] = load2(&Bs[n * LDB_S + 8 * half], &Bs[n * LDB_S + 16 + 8 * half]);
    }
    #pragma unroll
    for (int mt = 0; mt < 4; mt++)
      #pragma unroll
      for (int nt = 0; nt < 2; nt++)
        acc[mt][nt] = wmma_bf16(afr[mt], bfr[nt], acc[mt][nt]);
    __syncthreads();
  }

  // C layout: VGPR r holds row (r + 8*half), col = lane%16
  #pragma unroll
  for (int mt = 0; mt < 4; mt++)
    #pragma unroll
    for (int nt = 0; nt < 2; nt++)
      #pragma unroll
      for (int r = 0; r < 8; r++) {
        const int row = bRow + waveM * 64 + mt * 16 + r + 8 * half;
        const int col = bCol + waveN * 32 + nt * 16 + lc;
        C[(size_t)row * N + col] = acc[mt][nt][r];
      }
}

// ---------------- fused RMSNorm + RoPE, f32 -> bf16, head-major output ----------------
__global__ __launch_bounds__(128) void rms_rope_kernel(
    const float* __restrict__ in, const float* __restrict__ w,
    const float* __restrict__ cosT, const float* __restrict__ sinT,
    bf_t* __restrict__ out, int nheads, int rows_total, int row_off) {
  const int bid = blockIdx.x;
  const int t = bid / nheads, h = bid % nheads;
  const int d = threadIdx.x;
  __shared__ float red[HDs];
  __shared__ float nx[HDs];
  const float x = in[((size_t)t * nheads + h) * HDs + d];
  red[d] = x * x;
  __syncthreads();
  for (int s = HDs / 2; s > 0; s >>= 1) {
    if (d < s) red[d] += red[d + s];
    __syncthreads();
  }
  const float var = red[0] * (1.0f / HDs);
  const float nv = x * rsqrtf(var + 1e-6f) * w[d];
  nx[d] = nv;
  __syncthreads();
  const float rot = (d < HDs / 2) ? -nx[d + HDs / 2] : nx[d - HDs / 2];
  const float o = nv * cosT[(size_t)t * HDs + d] + rot * sinT[(size_t)t * HDs + d];
  out[((size_t)h * rows_total + row_off + t) * HDs + d] = (bf_t)o;
}

// ---------------- V pass-through into concatenated KV buffer ----------------
__global__ void vcopy_kernel(const float* __restrict__ vin, bf_t* __restrict__ vf) {
  const long n = (long)Tq * NKV * HDs;
  long idx = (long)blockIdx.x * blockDim.x + threadIdx.x;
  const long stride = (long)gridDim.x * blockDim.x;
  for (; idx < n; idx += stride) {
    const int t = (int)(idx / (NKV * HDs));
    const int rem = (int)(idx % (NKV * HDs));
    const int kvh = rem / HDs, d = rem % HDs;
    vf[((size_t)kvh * STOT + RV + t) * HDs + d] = (bf_t)vin[idx];
  }
}

// ---------------- alpha * virtual KV injection (rows 0..R-1) ----------------
__global__ void inject_kernel(const float* __restrict__ vk, const float* __restrict__ vv,
                              const float* __restrict__ alpha_logit,
                              bf_t* __restrict__ kf, bf_t* __restrict__ vf) {
  const float alpha = 0.5f / (1.0f + __expf(-alpha_logit[0]));   // MAX_ALPHA*sigmoid
  const long n = (long)NKV * RV * HDs;
  long idx = (long)blockIdx.x * blockDim.x + threadIdx.x;
  const long stride = (long)gridDim.x * blockDim.x;
  for (; idx < n; idx += stride) {
    const int kvh = (int)(idx / (RV * HDs));
    const int rem = (int)(idx % (RV * HDs));
    const int r = rem / HDs, d = rem % HDs;
    const size_t o = ((size_t)kvh * STOT + r) * HDs + d;
    kf[o] = (bf_t)(vk[idx] * alpha);
    vf[o] = (bf_t)(vv[idx] * alpha);
  }
}

// ---------------- flash attention (causal + virtual prefix) ----------------
// grid: (Tq/64, NH); 128 threads = 4 waves; wave w owns 16 query rows.
// K tile staged by the Tensor Data Mover (TENSORcnt path); wait overlaps V staging.
__global__ __launch_bounds__(128) void flash_kernel(
    const bf_t* __restrict__ Q,     // [NH][Tq][HD]
    const bf_t* __restrict__ Kf,    // [NKV][STOT][HD]
    const bf_t* __restrict__ Vf,    // [NKV][STOT][HD]
    bf_t* __restrict__ Oa) {        // [Tq][NH*HD]
  __shared__ alignas(16) bf_t Kl[32 * 136];          // [key][hd], stride 136 (=128+4DW pad)
  __shared__ alignas(16) bf_t Vt[128 * 40];          // transposed: [hd][key]
  __shared__ alignas(16) bf_t Pl[4][16 * 40];        // per-wave P strip [row][key]
  const int qb = blockIdx.x, h = blockIdx.y;
  const int kv = h / NREP;
  const int tid = threadIdx.x, lane = tid & 31, wave = tid >> 5;
  const int lc = lane & 15, half = lane >> 4;
  const int q0 = qb * 64;
  const int qrow_base = q0 + wave * 16;

  v16bf qfr[4];
  {
    const bf_t* qp = Q + ((size_t)h * Tq + qrow_base + lc) * HDs;
    #pragma unroll
    for (int c = 0; c < 4; c++)
      qfr[c] = load2(qp + 32 * c + 8 * half, qp + 32 * c + 16 + 8 * half);
  }
  v8f acc[8];
  #pragma unroll
  for (int j = 0; j < 8; j++) acc[j] = zero8();
  float mrow[8], lrow[8];
  #pragma unroll
  for (int r = 0; r < 8; r++) { mrow[r] = -3.0e38f; lrow[r] = 0.0f; }

  const int ktiles = (RV + q0 + 64) / 32;            // exact: all multiples of 32
  const float scale = 0.08838834764831845f;          // 1/sqrt(128)
  const int lkey = tid >> 2, lseg = tid & 3;
  const bf_t* kbaseptr = Kf + (size_t)kv * STOT * HDs;
  const bf_t* vbaseptr = Vf + (size_t)kv * STOT * HDs;

  // D# group1 (constant across tiles):
  //  data_size=2B(code 1), pad_enable, pad_interval=64DW(code 5), pad_amount=4DW(code 3)
  //  tensor_dim0=128 (bits79:48), tensor_dim1=STOT (bits111:80),
  //  tile_dim0=128 (bits127:112), tile_dim1=32 (bits143:128), dim0_stride=128 (bits207:160)
  int8v tdm_g1;
  tdm_g1[0] = (1 << 16) | (1 << 20) | (5 << 22) | (3 << 25);
  tdm_g1[1] = (int)(128u << 16);
  tdm_g1[2] = (int)((unsigned)STOT << 16);
  tdm_g1[3] = (int)(128u << 16);
  tdm_g1[4] = 32;
  tdm_g1[5] = 128;
  tdm_g1[6] = 0;
  tdm_g1[7] = 0;
  const unsigned kl_lds = (unsigned)(size_t)(void*)Kl;

  for (int kt = 0; kt < ktiles; kt++) {
    const int kb = kt * 32;
    if (wave == 0) {   // one TDM op per block per tile: 32x128 bf16 -> padded LDS rows
      unsigned long long ga =
          (unsigned long long)(size_t)(const void*)(kbaseptr + (size_t)kb * HDs);
      uint4v g0;
      g0[0] = 1u;                                          // count=1 valid descriptor
      g0[1] = kl_lds;                                      // lds_addr
      g0[2] = (unsigned)(ga & 0xFFFFFFFFull);              // global_addr[31:0]
      g0[3] = (unsigned)((ga >> 32) & 0x01FFFFFFull) | (2u << 30);  // addr[56:32]|type=2
      tdm_load(g0, tdm_g1);                                // TENSORcnt++
    }
    { // stage V tile 32x128, transposed (DMA cannot transpose) -- overlaps the TDM
      const bf_t* src = vbaseptr + (size_t)(kb + lkey) * HDs + lseg * 32;
      #pragma unroll
      for (int i = 0; i < 4; i++) {
        v8bf v = *(const v8bf*)(src + i * 8);
        #pragma unroll
        for (int e = 0; e < 8; e++)
          Vt[(lseg * 32 + i * 8 + e) * 40 + lkey] = v[e];
      }
    }
    if (wave == 0) wait_tensor0();                         // drain TDM before barrier
    __syncthreads();

    // S = Q K^T over 4 HD chunks, two 16-key column subtiles
    v8f s[2];
    #pragma unroll
    for (int nt = 0; nt < 2; nt++) {
      s[nt] = zero8();
      const int col = nt * 16 + lc;                  // key column = B lane col
      #pragma unroll
      for (int c = 0; c < 4; c++) {
        v16bf kfr = load2(&Kl[col * 136 + 32 * c + 8 * half],
                          &Kl[col * 136 + 32 * c + 16 + 8 * half]);
        s[nt] = wmma_bf16(qfr[c], kfr, s[nt]);
      }
    }

    // online softmax; row stats are per (r, half), reduced across the 16-lane half
    #pragma unroll
    for (int r = 0; r < 8; r++) {
      const int row = qrow_base + r + 8 * half;
      float s0 = s[0][r] * scale, s1 = s[1][r] * scale;
      const int j0 = kb + lc, j1 = kb + 16 + lc;
      if (j0 >= RV && (j0 - RV) > row) s0 = -1.0e30f;
      if (j1 >= RV && (j1 - RV) > row) s1 = -1.0e30f;
      float mt = fmaxf(s0, s1);
      for (int md = 1; md < 16; md <<= 1) mt = fmaxf(mt, __shfl_xor(mt, md, 32));
      const float mnew = fmaxf(mrow[r], mt);
      const float f = __expf(mrow[r] - mnew);
      mrow[r] = mnew;
      const float p0 = __expf(s0 - mnew);
      const float p1 = __expf(s1 - mnew);
      float ps = p0 + p1;
      for (int md = 1; md < 16; md <<= 1) ps += __shfl_xor(ps, md, 32);
      lrow[r] = lrow[r] * f + ps;
      #pragma unroll
      for (int j = 0; j < 8; j++) acc[j][r] *= f;
      Pl[wave][(r + 8 * half) * 40 + lc] = (bf_t)p0;
      Pl[wave][(r + 8 * half) * 40 + 16 + lc] = (bf_t)p1;
    }
    __syncthreads();

    // O += P(16x32) * V(32x128) in 8 hd tiles
    v16bf pfr = load2(&Pl[wave][lc * 40 + 8 * half],
                      &Pl[wave][lc * 40 + 16 + 8 * half]);
    #pragma unroll
    for (int j = 0; j < 8; j++) {
      v16bf vfr = load2(&Vt[(j * 16 + lc) * 40 + 8 * half],
                        &Vt[(j * 16 + lc) * 40 + 16 + 8 * half]);
      acc[j] = wmma_bf16(pfr, vfr, acc[j]);
    }
    __syncthreads();
  }

  #pragma unroll
  for (int j = 0; j < 8; j++)
    #pragma unroll
    for (int r = 0; r < 8; r++) {
      const int row = qrow_base + r + 8 * half;
      const int col = h * HDs + j * 16 + lc;
      Oa[(size_t)row * (NH * HDs) + col] = (bf_t)(acc[j][r] / lrow[r]);
    }
}

// ---------------- driver ----------------
extern "C" void kernel_launch(void* const* d_in, const int* in_sizes, int n_in,
                              void* d_out, int out_size, void* d_ws, size_t ws_size,
                              hipStream_t stream) {
  (void)in_sizes; (void)n_in; (void)out_size; (void)ws_size;
  const float* hidden = (const float*)d_in[0];
  // d_in[1] = attention_mask (causal structure reconstructed analytically)
  const float* cosT = (const float*)d_in[2];
  const float* sinT = (const float*)d_in[3];
  const float* vk   = (const float*)d_in[4];
  const float* vv   = (const float*)d_in[5];
  const float* Wq   = (const float*)d_in[6];
  const float* Wk   = (const float*)d_in[7];
  const float* Wv   = (const float*)d_in[8];
  const float* Wo   = (const float*)d_in[9];
  const float* qw   = (const float*)d_in[10];
  const float* kw   = (const float*)d_in[11];
  const float* alog = (const float*)d_in[12];
  float* out = (float*)d_out;

  char* ws = (char*)d_ws;
  size_t off = 0;
  auto take = [&](size_t bytes) -> char* {
    char* p = ws + off;
    off += (bytes + 255) & ~(size_t)255;
    return p;
  };
  bf_t* hid_b  = (bf_t*)take((size_t)Tq * Dm * 2);
  bf_t* Wq_b   = (bf_t*)take((size_t)Dm * (NH * HDs) * 2);
  bf_t* Wk_b   = (bf_t*)take((size_t)Dm * (NKV * HDs) * 2);
  bf_t* Wv_b   = (bf_t*)take((size_t)Dm * (NKV * HDs) * 2);
  bf_t* Wo_b   = (bf_t*)take((size_t)(NH * HDs) * Dm * 2);
  float* q_f   = (float*)take((size_t)Tq * (NH * HDs) * 4);
  float* k_f   = (float*)take((size_t)Tq * (NKV * HDs) * 4);
  float* v_f   = (float*)take((size_t)Tq * (NKV * HDs) * 4);
  bf_t* q_b    = (bf_t*)take((size_t)NH * Tq * HDs * 2);
  bf_t* k_full = (bf_t*)take((size_t)NKV * STOT * HDs * 2);
  bf_t* v_full = (bf_t*)take((size_t)NKV * STOT * HDs * 2);
  bf_t* attn_b = (bf_t*)take((size_t)Tq * (NH * HDs) * 2);

  // 1) casts to bf16
  cast_f32_bf16<<<4096, 256, 0, stream>>>(hidden, hid_b, (long)Tq * Dm);
  cast_f32_bf16<<<4096, 256, 0, stream>>>(Wq, Wq_b, (long)Dm * NH * HDs);
  cast_f32_bf16<<<4096, 256, 0, stream>>>(Wk, Wk_b, (long)Dm * NKV * HDs);
  cast_f32_bf16<<<4096, 256, 0, stream>>>(Wv, Wv_b, (long)Dm * NKV * HDs);
  cast_f32_bf16<<<4096, 256, 0, stream>>>(Wo, Wo_b, (long)NH * HDs * Dm);

  // 2) projections (WMMA GEMMs)
  gemm_bf16<<<dim3((NH * HDs) / 128, Tq / 128), 256, 0, stream>>>(hid_b, Wq_b, q_f, Tq, NH * HDs, Dm);
  gemm_bf16<<<dim3((NKV * HDs) / 128, Tq / 128), 256, 0, stream>>>(hid_b, Wk_b, k_f, Tq, NKV * HDs, Dm);
  gemm_bf16<<<dim3((NKV * HDs) / 128, Tq / 128), 256, 0, stream>>>(hid_b, Wv_b, v_f, Tq, NKV * HDs, Dm);

  // 3) RMSNorm + RoPE epilogues, V copy, virtual-KV injection
  rms_rope_kernel<<<Tq * NH, HDs, 0, stream>>>(q_f, qw, cosT, sinT, q_b, NH, Tq, 0);
  rms_rope_kernel<<<Tq * NKV, HDs, 0, stream>>>(k_f, kw, cosT, sinT, k_full, NKV, STOT, RV);
  vcopy_kernel<<<2048, 256, 0, stream>>>(v_f, v_full);
  inject_kernel<<<512, 256, 0, stream>>>(vk, vv, alog, k_full, v_full);

  // 4) flash attention (WMMA + TDM K-tile staging)
  flash_kernel<<<dim3(Tq / 64, NH), 128, 0, stream>>>(q_b, k_full, v_full, attn_b);

  // 5) output projection (WMMA GEMM) -> f32 out
  gemm_bf16<<<dim3(Dm / 128, Tq / 128), 256, 0, stream>>>(attn_b, Wo_b, out, Tq, Dm, NH * HDs);
}
